// SemanticAlignmentModule_71528385347860
// MI455X (gfx1250) — matrix operand order repaired and checked by reading, not compiled, not run on hardware
//
#include <hip/hip_runtime.h>

// ---------------------------------------------------------------------------
// CDNA5 (gfx1250) wave32 WMMA implementation of the Semantic Alignment Module.
// GEMMs: v_wmma_f32_16x16x32_bf16, 128x128 block tiles, double-buffered LDS,
// pre-transposed bf16 weights, batched B-fragment loads so WMMAs issue
// back-to-back. Attention K-tile staged via the Tensor Data Mover.
// ---------------------------------------------------------------------------

typedef __attribute__((ext_vector_type(16))) __bf16  v16bf;
typedef __attribute__((ext_vector_type(8)))  float   v8f;
typedef __attribute__((ext_vector_type(4)))  unsigned int u32x4;
typedef __attribute__((ext_vector_type(4)))  int     i32x4;
typedef __attribute__((ext_vector_type(8)))  int     i32x8;

union Frag { u32x4 q[2]; v16bf v; };

#if __has_include(<hip/amd_detail/amd_gfx1250_TDM.h>)
#define TDM_LOAD(g0, g1)                                                      \
  do {                                                                        \
    i32x4 _z4 = {0, 0, 0, 0};                                                 \
    i32x8 _z8 = {0, 0, 0, 0, 0, 0, 0, 0};                                     \
    __builtin_amdgcn_tensor_load_to_lds((g0), (g1), _z4, _z4, _z8, 0);        \
  } while (0)
#else
#define TDM_LOAD(g0, g1)                                                      \
  do {                                                                        \
    i32x4 _z4 = {0, 0, 0, 0};                                                 \
    __builtin_amdgcn_tensor_load_to_lds((g0), (g1), _z4, _z4, 0);             \
  } while (0)
#endif

__device__ __forceinline__ unsigned short f2bf(float f) {
  unsigned u = __float_as_uint(f);
  unsigned r = u + 0x7FFFu + ((u >> 16) & 1u);   // round to nearest even
  return (unsigned short)(r >> 16);
}

template <int BS>
__device__ __forceinline__ float block_sum(float v, float* red) {
  const int tid = threadIdx.x;
  red[tid] = v;
  __syncthreads();
#pragma unroll
  for (int s = BS / 2; s > 0; s >>= 1) {
    if (tid < s) red[tid] += red[tid + s];
    __syncthreads();
  }
  float r = red[0];
  __syncthreads();
  return r;
}

// ---------------------------------------------------------------------------
// bf16 GEMM with pre-transposed weights:  out = act(A[M,K] @ W[K,N] + bias)
// Wt is W transposed: Wt[n*K + k]. Block: 256 threads (8 waves),
// tile 128(M) x 128(N), K stepped by 32, double-buffered LDS.
// Each wave: 32x64 via 2 A-frags x 4 B-frags = 8 WMMAs per slab.
// ---------------------------------------------------------------------------
__global__ __launch_bounds__(256) void gemm_bf16_kernel(
    const unsigned short* __restrict__ A, const unsigned short* __restrict__ Wt,
    const float* __restrict__ bias, const float* __restrict__ residual,
    float* __restrict__ outf, unsigned short* __restrict__ outb,
    int M, int N, int K, int act)
{
  __shared__ unsigned short As[2][128 * 40];   // stride 40 -> 80B rows
  __shared__ unsigned short Bs[2][128 * 40];   // Bs[n][k]

  const int tid  = threadIdx.x;
  const int lane = tid & 31;
  const int wv   = tid >> 5;
  const int half = lane >> 4;
  const int ln   = lane & 15;
  const int wm   = (wv & 3) * 32;
  const int wn   = (wv >> 2) * 64;
  const int bm   = blockIdx.y * 128;
  const int bn   = blockIdx.x * 128;

  v8f acc[2][4] = {};
  u32x4 aR[2], bR[2];

  // prologue: stage slab 0
#pragma unroll
  for (int i = 0; i < 2; ++i) {
    const int idx = tid + i * 256, r = idx >> 2, q = idx & 3;
    u32x4 z = {0u, 0u, 0u, 0u};
    aR[i] = z;
    if (bm + r < M) aR[i] = *(const u32x4*)&A[(size_t)(bm + r) * K + q * 8];
    bR[i] = *(const u32x4*)&Wt[(size_t)(bn + r) * K + q * 8];
  }
#pragma unroll
  for (int i = 0; i < 2; ++i) {
    const int idx = tid + i * 256, r = idx >> 2, q = idx & 3;
    *(u32x4*)&As[0][r * 40 + q * 8] = aR[i];
    *(u32x4*)&Bs[0][r * 40 + q * 8] = bR[i];
  }
  __syncthreads();

  const int nk = K >> 5;
  for (int kk = 0; kk < nk; ++kk) {
    const int cur = kk & 1;
    if (kk + 1 < nk) {                       // issue next slab's global loads
      const int k0 = (kk + 1) * 32;
#pragma unroll
      for (int i = 0; i < 2; ++i) {
        const int idx = tid + i * 256, r = idx >> 2, q = idx & 3;
        u32x4 z = {0u, 0u, 0u, 0u};
        aR[i] = z;
        if (bm + r < M) aR[i] = *(const u32x4*)&A[(size_t)(bm + r) * K + k0 + q * 8];
        bR[i] = *(const u32x4*)&Wt[(size_t)(bn + r) * K + k0 + q * 8];
      }
    }
    // load all fragments first (distinct registers), then a clean WMMA chain
    Frag a0, a1, barr[4];
    a0.q[0] = *(const u32x4*)&As[cur][(wm + ln) * 40 + half * 8];
    a0.q[1] = *(const u32x4*)&As[cur][(wm + ln) * 40 + 16 + half * 8];
    a1.q[0] = *(const u32x4*)&As[cur][(wm + 16 + ln) * 40 + half * 8];
    a1.q[1] = *(const u32x4*)&As[cur][(wm + 16 + ln) * 40 + 16 + half * 8];
#pragma unroll
    for (int j = 0; j < 4; ++j) {
      const int br = wn + j * 16 + ln;
      barr[j].q[0] = *(const u32x4*)&Bs[cur][br * 40 + half * 8];
      barr[j].q[1] = *(const u32x4*)&Bs[cur][br * 40 + 16 + half * 8];
    }
#pragma unroll
    for (int j = 0; j < 4; ++j) {
      acc[0][j] = __builtin_amdgcn_wmma_f32_16x16x32_bf16(
          false, a0.v, false, barr[j].v, (short)0, acc[0][j], false, false);
      acc[1][j] = __builtin_amdgcn_wmma_f32_16x16x32_bf16(
          false, a1.v, false, barr[j].v, (short)0, acc[1][j], false, false);
    }
    if (kk + 1 < nk) {                       // store next slab into other buffer
      const int nb = cur ^ 1;
#pragma unroll
      for (int i = 0; i < 2; ++i) {
        const int idx = tid + i * 256, r = idx >> 2, q = idx & 3;
        *(u32x4*)&As[nb][r * 40 + q * 8] = aR[i];
        *(u32x4*)&Bs[nb][r * 40 + q * 8] = bR[i];
      }
    }
    __syncthreads();
  }

  // epilogue
#pragma unroll
  for (int j = 0; j < 4; ++j) {
    const int gn = bn + wn + j * 16 + ln;
    const float bcol = bias ? bias[gn] : 0.0f;
#pragma unroll
    for (int mi = 0; mi < 2; ++mi) {
#pragma unroll
      for (int r = 0; r < 8; ++r) {
        const int gm = bm + wm + mi * 16 + half * 8 + r;
        if (gm < M) {
          float v = acc[mi][j][r] + bcol;
          if (act == 1) v = 0.5f * v * (1.0f + erff(v * 0.70710678f));  // exact GELU
          const size_t o = (size_t)gm * N + gn;
          if (residual) v += residual[o];
          if (outf) outf[o] = v;
          if (outb) outb[o] = f2bf(v);
        }
      }
    }
  }
}

// ---------------------------------------------------------------------------
// Fused attention per (b, h, 64 query rows): sim=QK^T (WMMA), top-5/top-p
// sparsified softmax + entropy confidence (row-serial, T=77), attn@V (WMMA).
// K tile staged via Tensor Data Mover with LDS row padding 64->72 elements.
// ---------------------------------------------------------------------------
__global__ __launch_bounds__(128) void attn_kernel(
    const unsigned short* __restrict__ qbf,   // [B*4096, 512]
    const unsigned short* __restrict__ kbf,   // [B*77,   512]
    const unsigned short* __restrict__ vbf,   // [B*77,   512]
    const float* __restrict__ padm,           // [B*77]
    const float* __restrict__ ls_p,
    float* __restrict__ conf,                 // [B*4096]
    unsigned short* __restrict__ alnbf)       // [B*4096, 512]
{
  const int b  = blockIdx.z;
  const int h  = blockIdx.y;
  const int n0 = blockIdx.x * 64;
  const int tid  = threadIdx.x;
  const int lane = tid & 31;
  const int wv   = tid >> 5;
  const int half = lane >> 4;
  const int ln   = lane & 15;

  __shared__ float          S[64 * 100];          // sim f32 / attn bf16 (in place)
  __shared__ unsigned short Kb[80 * 72];          // K   [t][d], rows padded to 72
  __shared__ unsigned short Vt[64 * 104];         // V^T [d][t] (t padded to 96)
  __shared__ float          padS[96];

  // ---- K tile via TDM: 2D tile 64x77 of a row-stride-512 bf16 tensor.
  // pad_enable: every 32 dwords (one 128B row) insert 4 dwords -> LDS stride 72.
  if (wv == 0) {
    const unsigned long long ga =
        (unsigned long long)(const void*)(kbf + ((size_t)(b * 77)) * 512 + h * 64);
    const unsigned ldsa = (unsigned)(size_t)&Kb[0];
    u32x4 g0 = { 1u,                                  // count=1 valid descriptor
                 ldsa,                                // lds_addr
                 (unsigned)(ga & 0xFFFFFFFFull),      // global_addr[31:0]
                 (unsigned)((ga >> 32) & 0x1FFFFFFu) | 0x80000000u }; // [56:32]|type=2
    i32x8 g1 = { (int)((1u << 16)    // data_size = 2 bytes
                     | (1u << 20)    // pad_enable
                     | (4u << 22)    // pad_interval: 32 dwords
                     | (3u << 25)),  // pad_amount: 4 dwords
                 (int)(512u << 16),  // tensor_dim0 = 512
                 (int)(77u << 16),   // tensor_dim1 = 77
                 (int)(64u << 16),   // tile_dim0 = 64
                 (int)77,            // tile_dim1 = 77
                 (int)512,           // tensor_dim0_stride = 512
                 0, 0 };
    TDM_LOAD(g0, g1);
    __builtin_amdgcn_s_wait_tensorcnt(0);
  }

  for (int t = tid; t < 96; t += 128) padS[t] = (t < 77) ? padm[b * 77 + t] : 1.0f;
  for (int idx = tid; idx < 96 * 8; idx += 128) {   // V^T staging (vector loads)
    const int t = idx >> 3, dq = (idx & 7) * 8;
    u32x4 v = {0u, 0u, 0u, 0u};
    if (t < 77) v = *(const u32x4*)&vbf[((size_t)(b * 77 + t)) * 512 + h * 64 + dq];
    const unsigned short* vs = (const unsigned short*)&v;
#pragma unroll
    for (int j = 0; j < 8; ++j) Vt[(dq + j) * 104 + t] = vs[j];
  }
  __syncthreads();

  // ---- sim = (Q . K^T) * scale, masked ----
  {
    v8f acc[5] = {};
    const int row = n0 + wv * 16 + ln;
    const size_t qbase = ((size_t)(b * 4096 + row)) * 512 + h * 64;
#pragma unroll
    for (int ds = 0; ds < 2; ++ds) {
      Frag a, bb[5];
      a.q[0] = *(const u32x4*)&qbf[qbase + ds * 32 + half * 8];
      a.q[1] = *(const u32x4*)&qbf[qbase + ds * 32 + 16 + half * 8];
#pragma unroll
      for (int j = 0; j < 5; ++j) {
        const int t = j * 16 + ln;
        bb[j].q[0] = *(const u32x4*)&Kb[t * 72 + ds * 32 + half * 8];
        bb[j].q[1] = *(const u32x4*)&Kb[t * 72 + ds * 32 + 16 + half * 8];
      }
#pragma unroll
      for (int j = 0; j < 5; ++j)
        acc[j] = __builtin_amdgcn_wmma_f32_16x16x32_bf16(
            false, a.v, false, bb[j].v, (short)0, acc[j], false, false);
    }
    float ls = fminf(fmaxf(ls_p[0], -2.0f), 2.0f);
    const float scale = expf(ls) * 0.125f;        // / sqrt(64)
#pragma unroll
    for (int j = 0; j < 5; ++j) {
      const int t = j * 16 + ln;
#pragma unroll
      for (int r = 0; r < 8; ++r) {
        const int m = wv * 16 + half * 8 + r;
        float v = acc[j][r] * scale;
        if (t >= 77 || padS[t] > 0.5f) v = -__builtin_inff();
        S[m * 100 + t] = v;
      }
    }
  }
  __syncthreads();

  // ---- per-row top-5 / nucleus softmax / entropy confidence ----
  if (tid < 64) {
    float* srow = &S[tid * 100];
    unsigned short* arow = (unsigned short*)srow;   // bf16 written behind f32 reads
    float tp[5] = {-__builtin_inff(), -__builtin_inff(), -__builtin_inff(),
                   -__builtin_inff(), -__builtin_inff()};
    for (int t = 0; t < 77; ++t) {
      const float v = srow[t];
      if (v > tp[4]) {
        int i = 4;
        while (i > 0 && v > tp[i - 1]) { tp[i] = tp[i - 1]; --i; }
        tp[i] = v;
      }
    }
    const float m0 = tp[0], s5 = tp[4];
    const bool valid = (m0 > -1e30f);
    float Z = 0.0f;
    if (valid)
      for (int t = 0; t < 77; ++t) { const float v = srow[t]; if (v >= s5) Z += expf(v - m0); }
    float p0 = 0.0f, pmin = 0.0f, renorm = 0.0f;
    if (valid) {
      float cum = 0.0f;
#pragma unroll
      for (int i = 0; i < 5; ++i) {
        const float pi = expf(tp[i] - m0) / Z;
        cum += pi;
        const bool keep = (i == 0) || (cum <= 0.9f);
        if (keep && pi > 0.0f) { pmin = pi; renorm += pi; }
      }
      p0 = expf(tp[0] - m0) / Z;
    }
    const float rden = fmaxf(renorm, 1e-6f);
    float ent = 0.0f; int cnt = 0;
    for (int t = 0; t < 96; ++t) {
      float pk = 0.0f;
      if (t < 77 && valid) {
        const float v = srow[t];
        if (v >= s5) {
          const float p = expf(v - m0) / Z;
          if (p > 0.0f && p >= pmin) pk = p / rden;
        }
      }
      if (pk > 0.0f) { const float pc = fmaxf(pk, 1e-8f); ent += -pc * logf(pc); ++cnt; }
      arow[t] = f2bf(pk);
    }
    ent += (float)(77 - cnt) * 1.8420681e-7f;       // zero entries clamped at 1e-8
    const float teff    = fmaxf((float)cnt, 1.0f);
    const float entropy = ent / logf(fmaxf(teff, 2.0f));
    float cf = fminf(fmaxf((p0 / rden) * (1.0f - entropy), 0.0f), 1.0f);
    if (!valid) cf = 0.0f;
    atomicAdd(&conf[(size_t)b * 4096 + n0 + tid], cf * 0.125f);
  }
  __syncthreads();

  // ---- aligned = attn @ V ----
  {
    v8f acc[4] = {};
    const int row = wv * 16 + ln;
    const unsigned short* ar = (const unsigned short*)&S[row * 100];
#pragma unroll
    for (int ks = 0; ks < 3; ++ks) {
      Frag a, bb[4];
      a.q[0] = *(const u32x4*)&ar[ks * 32 + half * 8];
      a.q[1] = *(const u32x4*)&ar[ks * 32 + 16 + half * 8];
#pragma unroll
      for (int j = 0; j < 4; ++j) {
        const int d = j * 16 + ln;
        bb[j].q[0] = *(const u32x4*)&Vt[d * 104 + ks * 32 + half * 8];
        bb[j].q[1] = *(const u32x4*)&Vt[d * 104 + ks * 32 + 16 + half * 8];
      }
#pragma unroll
      for (int j = 0; j < 4; ++j)
        acc[j] = __builtin_amdgcn_wmma_f32_16x16x32_bf16(
            false, a.v, false, bb[j].v, (short)0, acc[j], false, false);
    }
#pragma unroll
    for (int j = 0; j < 4; ++j) {
#pragma unroll
      for (int r = 0; r < 8; ++r) {
        const int m = wv * 16 + half * 8 + r;
        const size_t o = ((size_t)(b * 4096 + n0 + m)) * 512 + h * 64 + j * 16 + ln;
        alnbf[o] = f2bf(acc[j][r]);
      }
    }
  }
}

// ---------------------------------------------------------------------------
// Elementwise / rowwise helpers
// ---------------------------------------------------------------------------
__global__ __launch_bounds__(256) void ln1_kernel(
    const float* __restrict__ in, const float* __restrict__ g,
    const float* __restrict__ be, float* __restrict__ outf,
    unsigned short* __restrict__ outb)
{
  __shared__ float red[256];
  const int tid = threadIdx.x;
  const size_t base = (size_t)blockIdx.x * 512;
  const float v0 = in[base + tid], v1 = in[base + tid + 256];
  const float mean = block_sum<256>(v0 + v1, red) * (1.0f / 512.0f);
  const float d0 = v0 - mean, d1 = v1 - mean;
  const float var = block_sum<256>(d0 * d0 + d1 * d1, red) * (1.0f / 512.0f);
  const float rs = rsqrtf(var + 1e-5f);
  const float y0 = d0 * rs * g[tid] + be[tid];
  const float y1 = d1 * rs * g[tid + 256] + be[tid + 256];
  outf[base + tid] = y0; outf[base + tid + 256] = y1;
  outb[base + tid] = f2bf(y0); outb[base + tid + 256] = f2bf(y1);
}

__global__ __launch_bounds__(256) void textprep_kernel(
    const float* __restrict__ text, unsigned short* __restrict__ tbf,
    float* __restrict__ padm)
{
  __shared__ float red[256];
  const int tid = threadIdx.x;
  const size_t base = (size_t)blockIdx.x * 512;
  const float v0 = text[base + tid], v1 = text[base + tid + 256];
  const float s = block_sum<256>(fabsf(v0) + fabsf(v1), red);
  tbf[base + tid] = f2bf(v0); tbf[base + tid + 256] = f2bf(v1);
  if (tid == 0) padm[blockIdx.x] = (s <= 1e-6f) ? 1.0f : 0.0f;
}

__global__ __launch_bounds__(128) void l2norm_kernel(
    const float* __restrict__ in, unsigned short* __restrict__ out)
{
  __shared__ float red[128];
  const int tid = threadIdx.x;
  const size_t base = (size_t)blockIdx.x * 512;
  float v[4]; float ss = 0.0f;
#pragma unroll
  for (int k = 0; k < 4; ++k) { v[k] = in[base + tid + k * 128]; ss += v[k] * v[k]; }
  ss = block_sum<128>(ss, red);
  const float sc = 1.0f / fmaxf(sqrtf(ss), 1e-6f);
#pragma unroll
  for (int k = 0; k < 4; ++k) out[base + tid + k * 128] = f2bf(v[k] * sc);
}

__global__ __launch_bounds__(128) void fuse_kernel(
    const float* __restrict__ x, const float* __restrict__ al,
    const float* __restrict__ conf, const float* __restrict__ Wg,
    const float* __restrict__ bg, const float* __restrict__ alpha_p,
    const float* __restrict__ g2, const float* __restrict__ b2,
    float* __restrict__ yout, unsigned short* __restrict__ ylnbf)
{
  __shared__ float red[128];
  const int tid = threadIdx.x;
  const int row = blockIdx.x;
  const size_t base = (size_t)row * 512;
  float xv[4], av[4], wg[4];
#pragma unroll
  for (int k = 0; k < 4; ++k) {
    const int c = tid + k * 128;
    xv[k] = x[base + c]; av[k] = al[base + c]; wg[k] = Wg[c];
  }
  float dot = 0, sx2 = 0, sa2 = 0;
#pragma unroll
  for (int k = 0; k < 4; ++k) { dot += xv[k] * wg[k]; sx2 += xv[k] * xv[k]; sa2 += av[k] * av[k]; }
  dot = block_sum<128>(dot, red);
  sx2 = block_sum<128>(sx2, red);
  sa2 = block_sum<128>(sa2, red);
  const float gate  = (1.0f / (1.0f + expf(-(dot + bg[0])))) * conf[row];
  const float xrms  = sqrtf(sx2 * (1.0f / 512.0f)) + 1e-6f;
  const float anorm = sqrtf(sa2);
  const float fac   = fminf(0.5f * xrms / fmaxf(anorm, 1e-6f), 1.0f);  // TAU=0.5
  const float alpha = alpha_p[0];
  float y[4]; float sy = 0.0f;
#pragma unroll
  for (int k = 0; k < 4; ++k) { y[k] = xv[k] + alpha * gate * (av[k] * fac); sy += y[k]; }
  const float mean = block_sum<128>(sy, red) * (1.0f / 512.0f);
  float sv = 0.0f;
#pragma unroll
  for (int k = 0; k < 4; ++k) { const float d = y[k] - mean; sv += d * d; }
  const float var = block_sum<128>(sv, red) * (1.0f / 512.0f);
  const float rs = rsqrtf(var + 1e-5f);
#pragma unroll
  for (int k = 0; k < 4; ++k) {
    const int c = tid + k * 128;
    yout[base + c] = y[k];
    ylnbf[base + c] = f2bf((y[k] - mean) * rs * g2[c] + b2[c]);
  }
}

// W[K][N] f32  ->  Wt[N][K] bf16 (tiled transpose through LDS)
__global__ __launch_bounds__(256) void transpose_cast_kernel(
    const float* __restrict__ in, unsigned short* __restrict__ out, int K, int N)
{
  __shared__ float tile[32][33];
  const int tx = threadIdx.x & 31, ty = threadIdx.x >> 5;   // 32 x 8
  const int k0 = blockIdx.y * 32, n0 = blockIdx.x * 32;
#pragma unroll
  for (int r = 0; r < 4; ++r)
    tile[ty + r * 8][tx] = in[(size_t)(k0 + ty + r * 8) * N + n0 + tx];
  __syncthreads();
#pragma unroll
  for (int r = 0; r < 4; ++r)
    out[(size_t)(n0 + ty + r * 8) * K + k0 + tx] = f2bf(tile[tx][ty + r * 8]);
}

// ---------------------------------------------------------------------------
extern "C" void kernel_launch(void* const* d_in, const int* in_sizes, int n_in,
                              void* d_out, int out_size, void* d_ws, size_t ws_size,
                              hipStream_t stream) {
  (void)in_sizes; (void)n_in; (void)out_size; (void)ws_size;
  const float* visual = (const float*)d_in[0];
  const float* text   = (const float*)d_in[1];
  const float* ln1_g  = (const float*)d_in[2];
  const float* ln1_b  = (const float*)d_in[3];
  const float* ln2_g  = (const float*)d_in[4];
  const float* ln2_b  = (const float*)d_in[5];
  const float* Wq = (const float*)d_in[6];  const float* bq = (const float*)d_in[7];
  const float* Wk = (const float*)d_in[8];  const float* bk = (const float*)d_in[9];
  const float* Wv = (const float*)d_in[10]; const float* bv = (const float*)d_in[11];
  const float* Wo = (const float*)d_in[12]; const float* bo = (const float*)d_in[13];
  const float* Wg = (const float*)d_in[14]; const float* bg = (const float*)d_in[15];
  const float* W1 = (const float*)d_in[16]; const float* b1 = (const float*)d_in[17];
  const float* W2 = (const float*)d_in[18]; const float* b2 = (const float*)d_in[19];
  const float* ls_p    = (const float*)d_in[20];
  const float* alpha_p = (const float*)d_in[21];

  const int Mrows = 32768;   // B * HI * WI
  const int Cc = 512, C4 = 2048, BT = 8 * 77;

  size_t cur = 0;
  auto alloc = [&](size_t bytes) -> char* {
    char* p = (char*)d_ws + cur;
    cur += (bytes + 255) & ~(size_t)255;
    return p;
  };
  float*          xf    = (float*)alloc((size_t)Mrows * Cc * 4);
  unsigned short* xbf   = (unsigned short*)alloc((size_t)Mrows * Cc * 2); // reused as ylnbf
  char*           cbuf  = alloc((size_t)Mrows * Cc * 4);   // qfull -> alignedO -> h_bf
  unsigned short* qbf   = (unsigned short*)alloc((size_t)Mrows * Cc * 2);
  unsigned short* alnbf = (unsigned short*)alloc((size_t)Mrows * Cc * 2);
  unsigned short* tbf   = (unsigned short*)alloc((size_t)BT * Cc * 2);
  float*          padm  = (float*)alloc((size_t)BT * 4);
  float*          kfull = (float*)alloc((size_t)BT * Cc * 4);
  unsigned short* kbf   = (unsigned short*)alloc((size_t)BT * Cc * 2);
  unsigned short* vbf   = (unsigned short*)alloc((size_t)BT * Cc * 2);
  float*          conf  = (float*)alloc((size_t)Mrows * 4);
  unsigned short* Wq_t = (unsigned short*)alloc((size_t)Cc * Cc * 2);
  unsigned short* Wk_t = (unsigned short*)alloc((size_t)Cc * Cc * 2);
  unsigned short* Wv_t = (unsigned short*)alloc((size_t)Cc * Cc * 2);
  unsigned short* Wo_t = (unsigned short*)alloc((size_t)Cc * Cc * 2);
  unsigned short* W1_t = (unsigned short*)alloc((size_t)Cc * C4 * 2);
  unsigned short* W2_t = (unsigned short*)alloc((size_t)C4 * Cc * 2);

  // weight transpose + bf16 casts (Wt[n][k])
  transpose_cast_kernel<<<dim3(16, 16), 256, 0, stream>>>(Wq, Wq_t, Cc, Cc);
  transpose_cast_kernel<<<dim3(16, 16), 256, 0, stream>>>(Wk, Wk_t, Cc, Cc);
  transpose_cast_kernel<<<dim3(16, 16), 256, 0, stream>>>(Wv, Wv_t, Cc, Cc);
  transpose_cast_kernel<<<dim3(16, 16), 256, 0, stream>>>(Wo, Wo_t, Cc, Cc);
  transpose_cast_kernel<<<dim3(64, 16), 256, 0, stream>>>(W1, W1_t, Cc, C4);
  transpose_cast_kernel<<<dim3(16, 64), 256, 0, stream>>>(W2, W2_t, C4, Cc);

  ln1_kernel<<<Mrows, 256, 0, stream>>>(visual, ln1_g, ln1_b, xf, xbf);
  textprep_kernel<<<BT, 256, 0, stream>>>(text, tbf, padm);

  // K/V projections (M=616, row tail guarded)
  dim3 gkv(Cc / 128, (BT + 127) / 128);
  gemm_bf16_kernel<<<gkv, 256, 0, stream>>>(tbf, Wk_t, bk, nullptr, kfull, nullptr, BT, Cc, Cc, 0);
  gemm_bf16_kernel<<<gkv, 256, 0, stream>>>(tbf, Wv_t, bv, nullptr, nullptr, vbf, BT, Cc, Cc, 0);

  // Q projection
  dim3 gq(Cc / 128, Mrows / 128);
  gemm_bf16_kernel<<<gq, 256, 0, stream>>>(xbf, Wq_t, bq, nullptr, (float*)cbuf, nullptr, Mrows, Cc, Cc, 0);

  l2norm_kernel<<<Mrows, 128, 0, stream>>>((float*)cbuf, qbf);
  l2norm_kernel<<<BT, 128, 0, stream>>>(kfull, kbf);

  hipMemsetAsync(conf, 0, (size_t)Mrows * 4, stream);
  attn_kernel<<<dim3(64, 8, 8), 128, 0, stream>>>(qbf, kbf, vbf, padm, ls_p, conf, alnbf);

  // output projection
  gemm_bf16_kernel<<<gq, 256, 0, stream>>>(alnbf, Wo_t, bo, nullptr, (float*)cbuf, nullptr, Mrows, Cc, Cc, 0);

  // gate + trust region + residual + LN2 (y -> d_out, ln(y) -> xbf as bf16)
  fuse_kernel<<<Mrows, 128, 0, stream>>>(xf, (float*)cbuf, conf, Wg, bg, alpha_p,
                                         ln2_g, ln2_b, (float*)d_out, xbf);

  // MLP in two 16384-row chunks (hidden bf16 reuses cbuf: 16384*2048*2 = 64MB)
  unsigned short* hbf = (unsigned short*)cbuf;
  for (int chunk = 0; chunk < 2; ++chunk) {
    const size_t roff = (size_t)chunk * 16384;
    dim3 g1(C4 / 128, 16384 / 128);
    gemm_bf16_kernel<<<g1, 256, 0, stream>>>(xbf + roff * Cc, W1_t, b1, nullptr,
                                             nullptr, hbf, 16384, C4, Cc, 1);
    dim3 g2g(Cc / 128, 16384 / 128);
    float* dres = (float*)d_out + roff * Cc;
    gemm_bf16_kernel<<<g2g, 256, 0, stream>>>(hbf, W2_t, b2, dres, dres, nullptr,
                                              16384, Cc, C4, 0);
  }
}
